// DecompositionModel_86182813762035
// MI455X (gfx1250) — matrix operand order repaired and checked by reading, not compiled, not run on hardware
//
#include <hip/hip_runtime.h>
#include <hip/hip_bf16.h>
#include <math.h>

// ---------------------------------------------------------------- types
typedef __attribute__((ext_vector_type(16))) __bf16 bf16x16;
typedef __attribute__((ext_vector_type(8)))  __bf16 bf16x8;
typedef __attribute__((ext_vector_type(8)))  float  v8f;
typedef __attribute__((ext_vector_type(4)))  unsigned int v4u;
typedef __attribute__((ext_vector_type(8)))  int    v8i;
typedef __attribute__((ext_vector_type(4)))  int    v4i;

#define ED_ 768
#define HD_ 48
#define NHEADS_ 16
#define BW_ 4
#define LQ_ 1024
#define LK_ 4096
#define XOUT_ELEMS (768*64*64)

// ---------------------------------------------------------------- helpers
__device__ inline v8f zero_v8f() { v8f z; for (int i = 0; i < 8; ++i) z[i] = 0.0f; return z; }
__device__ inline bf16x8 zero_b8() { bf16x8 z; for (int i = 0; i < 8; ++i) z[i] = (__bf16)0.0f; return z; }

__device__ inline v8f wmma_bf16(bf16x16 a, bf16x16 b, v8f c) {
    return __builtin_amdgcn_wmma_f32_16x16x32_bf16(false, a, false, b, (short)0, c, false, false);
}

// A operand: lane holds row (lane&15); contiguous 8-elem chunks at k0+g*8 and k0+16+g*8
__device__ inline bf16x16 load_opA(const __bf16* row, int k0, int kmax, int g) {
    int ka = k0 + g * 8;
    int kb = k0 + 16 + g * 8;
    bf16x8 lo = (ka < kmax) ? *(const bf16x8*)(row + ka) : zero_b8();
    bf16x8 hi = (kb < kmax) ? *(const bf16x8*)(row + kb) : zero_b8();
    bf16x16 r;
    for (int i = 0; i < 8; ++i) { r[i] = lo[i]; r[i + 8] = hi[i]; }
    return r;
}

// B operand: lane holds column (lane&15); contiguous 8-elem chunks at k0+g*8 and k0+16+g*8
__device__ inline bf16x16 load_opB(const __bf16* row, int k0, int kmax, int g) {
    int ka = k0 + g * 8;
    int kb = k0 + 16 + g * 8;
    bf16x8 lo = (ka < kmax) ? *(const bf16x8*)(row + ka) : zero_b8();
    bf16x8 hi = (kb < kmax) ? *(const bf16x8*)(row + kb) : zero_b8();
    bf16x16 r;
    for (int i = 0; i < 8; ++i) { r[i] = lo[i]; r[i + 8] = hi[i]; }
    return r;
}

// ---- Tensor Data Mover: 2D tile (bf16 elements) global -> LDS, tracked by TENSORcnt.
// D# per CDNA5 ISA: group0 {count=1, lds_addr, global_addr, type=2},
// group1 {data_size=1(2B), tensor_dim0/1 (large), tile_dim0/1, tensor_dim0_stride}.
__device__ inline void tdm_load_2d(void* lds_dst, const void* gsrc,
                                   unsigned tile_d0_elems, unsigned tile_d1,
                                   unsigned stride0_elems) {
    unsigned long long ga = (unsigned long long)(size_t)gsrc;
    v4u g0 = { 1u,                                        // count=1 (valid user D#)
               (unsigned)(size_t)lds_dst,                 // LDS byte address
               (unsigned)(ga & 0xFFFFFFFFull),            // global_addr[95:64]
               (unsigned)((ga >> 32) & 0x01FFFFFFull) | (2u << 30) };  // addr hi | type=2
    const unsigned td0 = 0x40000000u, td1 = 0x40000000u;  // tensor dims: large (tile fully in-bounds)
    v8i g1 = { (int)(1u << 16),                                             // data_size=1 -> 2 bytes
               (int)((td0 & 0xFFFFu) << 16),                                // tensor_dim0 lo
               (int)(((td0 >> 16) & 0xFFFFu) | ((td1 & 0xFFFFu) << 16)),    // dim0 hi | dim1 lo
               (int)(((td1 >> 16) & 0xFFFFu) | (tile_d0_elems << 16)),      // dim1 hi | tile_dim0
               (int)(tile_d1 & 0xFFFFu),                                    // tile_dim1 (tile_dim2=0)
               (int)stride0_elems,                                          // tensor_dim0_stride lo32
               0, 0 };
    v4i g2 = {0, 0, 0, 0};
    v4i g3 = {0, 0, 0, 0};
#if __clang_major__ >= 23
    v8i g4 = {0, 0, 0, 0, 0, 0, 0, 0};
    __builtin_amdgcn_tensor_load_to_lds(g0, g1, g2, g3, g4, 0);
#else
    __builtin_amdgcn_tensor_load_to_lds(g0, g1, g2, g3, 0);
#endif
}

// ---------------------------------------------------------------- K0: f32 -> bf16
__global__ void k_f32_to_bf16(const float* __restrict__ in, __bf16* __restrict__ out, int n) {
    int i = blockIdx.x * 256 + threadIdx.x;
    if (i < n) out[i] = (__bf16)in[i];
}

// ---------------------------------------------------------------- K1: bilinear resize (3,512,512)->(3,1024,1024)
__global__ void k_resize_bilinear(const float* __restrict__ x, float* __restrict__ rx) {
    int idx = blockIdx.x * 256 + threadIdx.x;
    if (idx >= 3 * 1024 * 1024) return;
    int c = idx >> 20;
    int rem = idx & ((1 << 20) - 1);
    int y = rem >> 10, xx = rem & 1023;
    float sy = fminf(fmaxf(y * 0.5f - 0.25f, 0.0f), 511.0f);
    float sx = fminf(fmaxf(xx * 0.5f - 0.25f, 0.0f), 511.0f);
    int y0 = (int)sy, x0 = (int)sx;
    int y1 = min(y0 + 1, 511), x1 = min(x0 + 1, 511);
    float wy = sy - y0, wx = sx - x0;
    const float* xc = x + (size_t)c * 512 * 512;
    float v00 = xc[y0 * 512 + x0], v01 = xc[y0 * 512 + x1];
    float v10 = xc[y1 * 512 + x0], v11 = xc[y1 * 512 + x1];
    rx[idx] = (v00 * (1 - wx) + v01 * wx) * (1 - wy) + (v10 * (1 - wx) + v11 * wx) * wy;
}

// ---------------------------------------------------------------- K2: origin conv 4x4/s4 grouped + LN -> bf16 (Bw,4096,768)
__global__ __launch_bounds__(256) void k_conv_origin_ln(
        const float* __restrict__ x, const float* __restrict__ wgt, const float* __restrict__ bias,
        const float* __restrict__ ln_g, const float* __restrict__ ln_b, __bf16* __restrict__ outbf) {
    int w = blockIdx.y, tok = blockIdx.x;
    int oy = tok >> 6, ox = tok & 63;
    int wi = w >> 1, wj = w & 1;
    __shared__ float patch[48];
    __shared__ float rs[256], rq[256];
    int tid = threadIdx.x;
    if (tid < 48) {
        int c = tid >> 4, p = tid & 15, ky = p >> 2, kx = p & 3;
        patch[tid] = x[((size_t)c * 512 + (wi * 256 + oy * 4 + ky)) * 512 + (wj * 256 + ox * 4 + kx)];
    }
    __syncthreads();
    float vals[3], lsum = 0.f, lsq = 0.f;
    for (int r = 0; r < 3; ++r) {
        int co = tid + r * 256;
        int gi = co >> 8;
        float acc = bias[co];
        for (int p = 0; p < 16; ++p) acc += patch[gi * 16 + p] * wgt[co * 16 + p];
        vals[r] = acc; lsum += acc; lsq += acc * acc;
    }
    rs[tid] = lsum; rq[tid] = lsq; __syncthreads();
    for (int s = 128; s > 0; s >>= 1) {
        if (tid < s) { rs[tid] += rs[tid + s]; rq[tid] += rq[tid + s]; }
        __syncthreads();
    }
    float mu = rs[0] / 768.f;
    float rstd = rsqrtf(rq[0] / 768.f - mu * mu + 1e-5f);
    for (int r = 0; r < 3; ++r) {
        int co = tid + r * 256;
        outbf[((size_t)w * 4096 + tok) * 768 + co] = (__bf16)((vals[r] - mu) * rstd * ln_g[co] + ln_b[co]);
    }
}

// ---------------------------------------------------------------- K3: resize conv 16x16/s16 grouped + pos + LN -> bf16 (Bw,1024,768)
__global__ __launch_bounds__(256) void k_conv_resize_ln(
        const float* __restrict__ rx, const float* __restrict__ wgt, const float* __restrict__ bias,
        const float* __restrict__ pos_embed, const float* __restrict__ ln_g, const float* __restrict__ ln_b,
        __bf16* __restrict__ outbf) {
    int w = blockIdx.y, tok = blockIdx.x;
    int ry = tok >> 5, rc = tok & 31;
    int wi = w >> 1, wj = w & 1;
    __shared__ float patch[768];
    __shared__ float rs[256], rq[256];
    int tid = threadIdx.x;
    for (int r = 0; r < 3; ++r) {
        int i = tid + r * 256;
        int c = i >> 8, p = i & 255, ky = p >> 4, kx = p & 15;
        patch[i] = rx[((size_t)c * 1024 + (wi * 512 + ry * 16 + ky)) * 1024 + (wj * 512 + rc * 16 + kx)];
    }
    __syncthreads();
    float vals[3], lsum = 0.f, lsq = 0.f;
    for (int r = 0; r < 3; ++r) {
        int co = tid + r * 256;
        int gi = co >> 8;
        float acc = bias[co];
        for (int p = 0; p < 256; ++p) acc += patch[gi * 256 + p] * wgt[co * 256 + p];
        acc += pos_embed[(size_t)tok * 768 + co];
        vals[r] = acc; lsum += acc; lsq += acc * acc;
    }
    rs[tid] = lsum; rq[tid] = lsq; __syncthreads();
    for (int s = 128; s > 0; s >>= 1) {
        if (tid < s) { rs[tid] += rs[tid + s]; rq[tid] += rq[tid + s]; }
        __syncthreads();
    }
    float mu = rs[0] / 768.f;
    float rstd = rsqrtf(rq[0] / 768.f - mu * mu + 1e-5f);
    for (int r = 0; r < 3; ++r) {
        int co = tid + r * 256;
        outbf[((size_t)w * 1024 + tok) * 768 + co] = (__bf16)((vals[r] - mu) * rstd * ln_g[co] + ln_b[co]);
    }
}

// ---------------------------------------------------------------- K4: WMMA GEMM C = A @ W^T + bias, head-scatter.
// W tile (64 N-rows x 128 K) is identical for all 8 waves -> TDM double-buffered into LDS.
__global__ __launch_bounds__(256) void k_gemm_headscatter(
        const __bf16* __restrict__ A, const __bf16* __restrict__ W,
        const float* __restrict__ bias, float* __restrict__ Out, int L) {
    __shared__ __bf16 sW[2][64 * 128];
    int lane = threadIdx.x & 31, wv = threadIdx.x >> 5;
    int g = lane >> 4, n0 = lane & 15;
    int w = blockIdx.z;
    int mbase = blockIdx.x * 128 + wv * 16;
    int nbase = blockIdx.y * 64;
    const __bf16* Arow = A + ((size_t)w * L + (mbase + n0)) * ED_;
    v8f acc[4];
    for (int t = 0; t < 4; ++t) acc[t] = zero_v8f();
    if (wv == 0)
        tdm_load_2d(&sW[0][0], W + (size_t)nbase * ED_, 128, 64, ED_);
    const int NCH = ED_ / 128;  // 6
#pragma unroll 1
    for (int c = 0; c < NCH; ++c) {
        if (wv == 0) __builtin_amdgcn_s_wait_tensorcnt(0);
        __syncthreads();  // tile c visible to all; buffer (c+1)&1 idle since chunk c-1
        if (wv == 0 && c + 1 < NCH)
            tdm_load_2d(&sW[(c + 1) & 1][0], W + (size_t)nbase * ED_ + (c + 1) * 128, 128, 64, ED_);
        const __bf16* Wl = &sW[c & 1][0];
        for (int k0l = 0; k0l < 128; k0l += 32) {
            bf16x16 a  = load_opA(Arow, c * 128 + k0l, ED_, g);
            bf16x16 b0 = load_opB(Wl + (0 * 16 + n0) * 128, k0l, 128, g);
            bf16x16 b1 = load_opB(Wl + (1 * 16 + n0) * 128, k0l, 128, g);
            bf16x16 b2 = load_opB(Wl + (2 * 16 + n0) * 128, k0l, 128, g);
            bf16x16 b3 = load_opB(Wl + (3 * 16 + n0) * 128, k0l, 128, g);
            acc[0] = wmma_bf16(a, b0, acc[0]);
            acc[1] = wmma_bf16(a, b1, acc[1]);
            acc[2] = wmma_bf16(a, b2, acc[2]);
            acc[3] = wmma_bf16(a, b3, acc[3]);
        }
    }
    for (int nt = 0; nt < 4; ++nt) {
        int n = nbase + nt * 16 + n0;
        int head = n / HD_, hd = n % HD_;
        float bi = bias[n];
        for (int v = 0; v < 8; ++v) {
            int m = mbase + v + 8 * g;
            Out[(((size_t)w * NHEADS_ + head) * L + m) * HD_ + hd] = acc[nt][v] + bi;
        }
    }
}

// ---------------------------------------------------------------- K5a: gate + l2norm + RoPE for q -> bf16 (Bw,16,1024,48)
__global__ void k_rope_q(const float* __restrict__ qbuf, const float* __restrict__ pos,
                         const float* __restrict__ qa_w, const float* __restrict__ qa_b,
                         __bf16* __restrict__ qrot) {
    int idx = blockIdx.x * 256 + threadIdx.x;
    if (idx >= BW_ * NHEADS_ * LQ_) return;
    int lq = idx & (LQ_ - 1);
    const float* row = qbuf + (size_t)idx * HD_;
    const float* prow = pos + (size_t)idx * HD_;
    float v[HD_];
    float dot = qa_b[0];
    for (int d = 0; d < HD_; ++d) { v[d] = row[d]; dot += row[d] * qa_w[d]; }
    float a = 1.0f / (1.0f + expf(-dot));
    float nrm = 0.f;
    for (int d = 0; d < HD_; ++d) { v[d] = v[d] * a + prow[d] * (1.0f - a); nrm += v[d] * v[d]; }
    float inv = 1.0f / fmaxf(sqrtf(nrm), 1e-12f);
    float tx = (float)(lq & 31), ty = (float)(lq >> 5);
    __bf16* orow = qrot + (size_t)idx * HD_;
    for (int p = 0; p < 24; ++p) {
        int j = (p < 12) ? p : (p - 12);
        float ang = ((p < 12) ? tx : ty) * powf(10.0f, -(float)j / 12.0f);
        float cs = cosf(ang), sn = sinf(ang);
        float xe = v[2 * p] * inv, xo = v[2 * p + 1] * inv;
        orow[2 * p]     = (__bf16)(xe * cs - xo * sn);
        orow[2 * p + 1] = (__bf16)(xe * sn + xo * cs);
    }
}

// ---------------------------------------------------------------- K5b: l2norm + RoPE for b -> bf16; also tmp_b^T bf16 (Bw,16,48,4096)
__global__ void k_rope_b(const float* __restrict__ bbuf, __bf16* __restrict__ brot,
                         __bf16* __restrict__ tbT) {
    int idx = blockIdx.x * 256 + threadIdx.x;
    if (idx >= BW_ * NHEADS_ * LK_) return;
    int lk = idx & (LK_ - 1);
    int wh = idx >> 12;  // w*16+h
    const float* row = bbuf + (size_t)idx * HD_;
    float v[HD_], nrm = 0.f;
    for (int d = 0; d < HD_; ++d) { v[d] = row[d]; nrm += v[d] * v[d]; }
    float inv = 1.0f / fmaxf(sqrtf(nrm), 1e-12f);
    for (int d = 0; d < HD_; ++d)
        tbT[((size_t)wh * HD_ + d) * LK_ + lk] = (__bf16)v[d];
    float tx = (float)(lk & 63), ty = (float)(lk >> 6);
    __bf16* orow = brot + (size_t)idx * HD_;
    for (int p = 0; p < 24; ++p) {
        int j = (p < 12) ? p : (p - 12);
        float ang = ((p < 12) ? tx : ty) * powf(10.0f, -(float)j / 12.0f);
        float cs = cosf(ang), sn = sinf(ang);
        float xe = v[2 * p] * inv, xo = v[2 * p + 1] * inv;
        orow[2 * p]     = (__bf16)(xe * cs - xo * sn);
        orow[2 * p + 1] = (__bf16)(xe * sn + xo * cs);
    }
}

// ---------------------------------------------------------------- K6: fused attention.
// Per 128-token chunk: TDM stages b_rot (128x48) and tmp_b^T (48x128) into LDS (shared by
// all 8 waves, double-buffered). S = q b^T stored to attn_map and fed through per-wave LDS
// into the A-operand of O += S @ tmp_b.
#define CHUNK_ 128
__global__ __launch_bounds__(256) void k_attention(
        const __bf16* __restrict__ qrot, const __bf16* __restrict__ brot,
        const __bf16* __restrict__ tbT, float* __restrict__ attn, __bf16* __restrict__ outbf) {
    __shared__ __bf16 sB[2][CHUNK_ * HD_];   // [token][dim]
    __shared__ __bf16 sT[2][HD_ * CHUNK_];   // [hd][token]
    __shared__ __bf16 sS[8][16 * 32 + 8];    // per-wave S tile in A-feed layout
    int lane = threadIdx.x & 31, wv = threadIdx.x >> 5;
    int g = lane >> 4, n0 = lane & 15;
    int w = blockIdx.z, h = blockIdx.y;
    int qbase = blockIdx.x * 128 + wv * 16;
    size_t headoff = (size_t)w * NHEADS_ + h;
    const __bf16* qrow = qrot + (headoff * LQ_ + qbase + n0) * HD_;
    const __bf16* bbase = brot + headoff * LK_ * HD_;
    const __bf16* tbase = tbT + headoff * HD_ * (size_t)LK_;
    bf16x16 aq0 = load_opA(qrow, 0, HD_, g);
    bf16x16 aq1 = load_opA(qrow, 32, HD_, g);
    v8f o[3];
    for (int t = 0; t < 3; ++t) o[t] = zero_v8f();
    float* attnbase = attn + headoff * LQ_ * (size_t)LK_;
    __bf16* lds = &sS[wv][0];
    if (wv == 0) {
        tdm_load_2d(&sB[0][0], bbase, CHUNK_ * HD_, 1, CHUNK_ * HD_);   // contiguous rows
        tdm_load_2d(&sT[0][0], tbase, CHUNK_, HD_, LK_);                // 48 rows, stride 4096
    }
    const int NCH = LK_ / CHUNK_;  // 32
#pragma unroll 1
    for (int c = 0; c < NCH; ++c) {
        if (wv == 0) __builtin_amdgcn_s_wait_tensorcnt(0);
        __syncthreads();
        if (wv == 0 && c + 1 < NCH) {
            int nb = (c + 1) & 1;
            tdm_load_2d(&sB[nb][0], bbase + (size_t)(c + 1) * CHUNK_ * HD_, CHUNK_ * HD_, 1, CHUNK_ * HD_);
            tdm_load_2d(&sT[nb][0], tbase + (size_t)(c + 1) * CHUNK_, CHUNK_, HD_, LK_);
        }
        const __bf16* Lb = &sB[c & 1][0];
        const __bf16* Lt = &sT[c & 1][0];
#pragma unroll 1
        for (int it = 0; it < CHUNK_ / 32; ++it) {
            int tl = it * 32;                 // token offset within chunk
            int ktg = c * CHUNK_ + tl;        // global token offset
            // --- QK^T: two 16x16 S tiles covering 32 b-tokens (B operand from LDS)
            for (int t = 0; t < 2; ++t) {
                const __bf16* brow = Lb + (tl + t * 16 + n0) * HD_;
                bf16x16 b0 = load_opB(brow, 0, HD_, g);
                bf16x16 b1 = load_opB(brow, 32, HD_, g);
                v8f s = zero_v8f();
                s = wmma_bf16(aq0, b0, s);
                s = wmma_bf16(aq1, b1, s);
                for (int v = 0; v < 8; ++v) {
                    int m = v + 8 * g;
                    attnbase[(size_t)(qbase + m) * LK_ + ktg + t * 16 + n0] = s[v];
                    lds[m * 32 + t * 16 + n0] = (__bf16)s[v];
                }
            }
            // --- A-operand of S from per-wave LDS (row = lane&15)
            bf16x16 aS;
            {
                bf16x8 lo = *(const bf16x8*)(lds + n0 * 32 + g * 8);
                bf16x8 hi = *(const bf16x8*)(lds + n0 * 32 + 16 + g * 8);
                for (int i = 0; i < 8; ++i) { aS[i] = lo[i]; aS[i + 8] = hi[i]; }
            }
            // --- O += S @ tmp_b (N=48 -> 3 tiles); B columns are rows of staged tmp_b^T
            for (int nt = 0; nt < 3; ++nt) {
                bf16x16 bV = load_opB(Lt + (nt * 16 + n0) * CHUNK_, tl, CHUNK_, g);
                o[nt] = wmma_bf16(aS, bV, o[nt]);
            }
        }
    }
    for (int nt = 0; nt < 3; ++nt)
        for (int v = 0; v < 8; ++v) {
            int m = qbase + v + 8 * g;
            outbf[((size_t)w * LQ_ + m) * ED_ + h * HD_ + nt * 16 + n0] = (__bf16)o[nt][v];
        }
}

// ---------------------------------------------------------------- K7: proj GEMM with window re-assembly; out (768,64,64)
__global__ __launch_bounds__(256) void k_gemm_proj(
        const __bf16* __restrict__ Abf, const __bf16* __restrict__ W,
        const float* __restrict__ bias, float* __restrict__ xout) {
    __shared__ __bf16 sW[2][64 * 128];
    int lane = threadIdx.x & 31, wv = threadIdx.x >> 5;
    int g = lane >> 4, n0 = lane & 15;
    int mbase = blockIdx.x * 128 + wv * 16;
    int nbase = blockIdx.y * 64;
    int mt = mbase + n0;
    int R = mt >> 6, C = mt & 63;
    int w = ((R >> 5) << 1) | (C >> 5);
    int lq = ((R & 31) << 5) | (C & 31);
    const __bf16* Arow = Abf + ((size_t)w * LQ_ + lq) * ED_;
    v8f acc[4];
    for (int t = 0; t < 4; ++t) acc[t] = zero_v8f();
    if (wv == 0)
        tdm_load_2d(&sW[0][0], W + (size_t)nbase * ED_, 128, 64, ED_);
    const int NCH = ED_ / 128;
#pragma unroll 1
    for (int c = 0; c < NCH; ++c) {
        if (wv == 0) __builtin_amdgcn_s_wait_tensorcnt(0);
        __syncthreads();
        if (wv == 0 && c + 1 < NCH)
            tdm_load_2d(&sW[(c + 1) & 1][0], W + (size_t)nbase * ED_ + (c + 1) * 128, 128, 64, ED_);
        const __bf16* Wl = &sW[c & 1][0];
        for (int k0l = 0; k0l < 128; k0l += 32) {
            bf16x16 a  = load_opA(Arow, c * 128 + k0l, ED_, g);
            bf16x16 b0 = load_opB(Wl + (0 * 16 + n0) * 128, k0l, 128, g);
            bf16x16 b1 = load_opB(Wl + (1 * 16 + n0) * 128, k0l, 128, g);
            bf16x16 b2 = load_opB(Wl + (2 * 16 + n0) * 128, k0l, 128, g);
            bf16x16 b3 = load_opB(Wl + (3 * 16 + n0) * 128, k0l, 128, g);
            acc[0] = wmma_bf16(a, b0, acc[0]);
            acc[1] = wmma_bf16(a, b1, acc[1]);
            acc[2] = wmma_bf16(a, b2, acc[2]);
            acc[3] = wmma_bf16(a, b3, acc[3]);
        }
    }
    for (int nt = 0; nt < 4; ++nt) {
        int n = nbase + nt * 16 + n0;
        float bi = bias[n];
        for (int v = 0; v < 8; ++v) {
            int m = mbase + v + 8 * g;
            xout[(size_t)n * 4096 + m] = acc[nt][v] + bi;
        }
    }
}

// ---------------------------------------------------------------- launcher
extern "C" void kernel_launch(void* const* d_in, const int* in_sizes, int n_in,
                              void* d_out, int out_size, void* d_ws, size_t ws_size,
                              hipStream_t stream) {
    const float* x        = (const float*)d_in[0];
    const float* conv_o_w = (const float*)d_in[1];
    const float* conv_o_b = (const float*)d_in[2];
    const float* conv_r_w = (const float*)d_in[3];
    const float* conv_r_b = (const float*)d_in[4];
    const float* b_w      = (const float*)d_in[5];
    const float* b_b      = (const float*)d_in[6];
    const float* q_w      = (const float*)d_in[7];
    const float* q_b      = (const float*)d_in[8];
    const float* pos_emb  = (const float*)d_in[9];
    const float* pos      = (const float*)d_in[10];
    const float* lnb_g    = (const float*)d_in[11];
    const float* lnb_b    = (const float*)d_in[12];
    const float* lnq_g    = (const float*)d_in[13];
    const float* lnq_b    = (const float*)d_in[14];
    const float* qa_w     = (const float*)d_in[15];
    const float* qa_b     = (const float*)d_in[16];
    const float* proj_w   = (const float*)d_in[17];
    const float* proj_b   = (const float*)d_in[18];

    size_t off = 0;
    char* wsb = (char*)d_ws;
    auto alloc = [&](size_t bytes) -> void* {
        void* p = wsb + off;
        off += (bytes + 255) & ~(size_t)255;
        return p;
    };
    float*  rx        = (float*) alloc((size_t)3 * 1024 * 1024 * 4);
    __bf16* origin_bf = (__bf16*)alloc((size_t)BW_ * 4096 * ED_ * 2);
    __bf16* resize_bf = (__bf16*)alloc((size_t)BW_ * 1024 * ED_ * 2);
    __bf16* qw_bf     = (__bf16*)alloc((size_t)ED_ * ED_ * 2);
    __bf16* bw_bf     = (__bf16*)alloc((size_t)ED_ * ED_ * 2);
    __bf16* pw_bf     = (__bf16*)alloc((size_t)ED_ * ED_ * 2);
    float*  qbuf      = (float*) alloc((size_t)BW_ * NHEADS_ * LQ_ * HD_ * 4);
    float*  bbuf      = (float*) alloc((size_t)BW_ * NHEADS_ * LK_ * HD_ * 4);
    __bf16* qrot      = (__bf16*)alloc((size_t)BW_ * NHEADS_ * LQ_ * HD_ * 2);
    __bf16* brot      = (__bf16*)alloc((size_t)BW_ * NHEADS_ * LK_ * HD_ * 2);
    __bf16* tbT       = (__bf16*)alloc((size_t)BW_ * NHEADS_ * HD_ * LK_ * 2);
    __bf16* outbf     = (__bf16*)alloc((size_t)BW_ * LQ_ * ED_ * 2);

    float* xout = (float*)d_out;
    float* attn = (float*)d_out + XOUT_ELEMS;

    const int WN = ED_ * ED_;
    k_f32_to_bf16<<<(WN + 255) / 256, 256, 0, stream>>>(q_w, qw_bf, WN);
    k_f32_to_bf16<<<(WN + 255) / 256, 256, 0, stream>>>(b_w, bw_bf, WN);
    k_f32_to_bf16<<<(WN + 255) / 256, 256, 0, stream>>>(proj_w, pw_bf, WN);

    k_resize_bilinear<<<(3 * 1024 * 1024) / 256, 256, 0, stream>>>(x, rx);

    k_conv_origin_ln<<<dim3(4096, BW_), 256, 0, stream>>>(x, conv_o_w, conv_o_b, lnb_g, lnb_b, origin_bf);
    k_conv_resize_ln<<<dim3(1024, BW_), 256, 0, stream>>>(rx, conv_r_w, conv_r_b, pos_emb, lnq_g, lnq_b, resize_bf);

    k_gemm_headscatter<<<dim3(LQ_ / 128, ED_ / 64, BW_), 256, 0, stream>>>(resize_bf, qw_bf, q_b, qbuf, LQ_);
    k_gemm_headscatter<<<dim3(LK_ / 128, ED_ / 64, BW_), 256, 0, stream>>>(origin_bf, bw_bf, b_b, bbuf, LK_);

    k_rope_q<<<(BW_ * NHEADS_ * LQ_) / 256, 256, 0, stream>>>(qbuf, pos, qa_w, qa_b, qrot);
    k_rope_b<<<(BW_ * NHEADS_ * LK_) / 256, 256, 0, stream>>>(bbuf, brot, tbT);

    k_attention<<<dim3(LQ_ / 128, NHEADS_, BW_), 256, 0, stream>>>(qrot, brot, tbT, attn, outbf);

    k_gemm_proj<<<dim3(4096 / 128, ED_ / 64, 1), 256, 0, stream>>>(outbf, pw_bf, proj_b, xout);
}
